// PointTransformerCLS_11982958756084
// MI455X (gfx1250) — compile-verified, matrix operand order checked by reading, not run on hardware
//
#include <hip/hip_runtime.h>

// PointTransformerCLS forward for gfx1250 (MI455X).
// All dense GEMMs run through v_wmma_f32_16x16x32_f16 (f32 inputs converted to
// f16 in-register at fragment load, f32 accumulate). Each wave computes a
// 32x32 C macro-tile (2x2 WMMA tiles) so every fragment feeds two WMMAs.
// Irregular ops (FPS, kNN, softmax, BN stats, gathers, pools) are wave32
// VALU/LDS kernels. Workspace requirement: ~740 MB.

typedef __attribute__((ext_vector_type(16))) _Float16 v16h;
typedef __attribute__((ext_vector_type(8)))  float    v8f;

#define TPB 256

// ---------------------------------------------------------------------------
// WMMA fragment load: A (row idx) or B (col idx) operand for 16x16x32 f16.
// 16-bit A 16x32 layout (ISA 7.12.2): lane L (row/col = L%16), kb = L<16?0:8,
// elements 0..7 -> K = k0+kb+0..7, elements 8..15 -> K = k0+kb+16..23.
// Requires (k0+kb) and row stride multiples of 4 floats (true everywhere here).
// ---------------------------------------------------------------------------
__device__ __forceinline__ v16h frag_load_f32(const float* base, int ld, int idx,
                                              int k0, int lane) {
  int kb = (lane < 16) ? 0 : 8;
  const float* p = base + (long long)idx * ld + k0 + kb;
  float4 f0 = *reinterpret_cast<const float4*>(p);
  float4 f1 = *reinterpret_cast<const float4*>(p + 4);
  float4 f2 = *reinterpret_cast<const float4*>(p + 16);
  float4 f3 = *reinterpret_cast<const float4*>(p + 20);
  v16h r;
  r[0]  = (_Float16)f0.x; r[1]  = (_Float16)f0.y; r[2]  = (_Float16)f0.z; r[3]  = (_Float16)f0.w;
  r[4]  = (_Float16)f1.x; r[5]  = (_Float16)f1.y; r[6]  = (_Float16)f1.z; r[7]  = (_Float16)f1.w;
  r[8]  = (_Float16)f2.x; r[9]  = (_Float16)f2.y; r[10] = (_Float16)f2.z; r[11] = (_Float16)f2.w;
  r[12] = (_Float16)f3.x; r[13] = (_Float16)f3.y; r[14] = (_Float16)f3.z; r[15] = (_Float16)f3.w;
  return r;
}

// Batched NT GEMM: C[m,n] = sum_k A[m,k]*B[n,k] (+bias). One wave = one 32x32
// macro-tile of C (2x2 WMMA tiles). M%32==0, N%32==0, K%32==0.
// biasMode: 0 none, 1 per-col, 2 per-row.
__global__ void wmma_gemm_kernel(const float* __restrict__ A, const float* __restrict__ B,
                                 float* __restrict__ C, int M, int N, int K,
                                 long long sA, long long sB, long long sC,
                                 const float* __restrict__ bias, int biasMode) {
  int lane = threadIdx.x & 31;
  int wave = threadIdx.x >> 5;
  long long tilesN = N >> 5;
  long long nTiles = (long long)(M >> 5) * tilesN;
  long long tile = (long long)blockIdx.x * 8 + wave;   // wave-uniform
  if (tile >= nTiles) return;                          // whole-wave exit, EXEC stays all-ones
  int tm = (int)(tile / tilesN);
  int tn = (int)(tile % tilesN);
  A += (long long)blockIdx.z * sA;
  B += (long long)blockIdx.z * sB;
  C += (long long)blockIdx.z * sC;
  int lm = lane & 15;
  int row0 = tm * 32, col0 = tn * 32;
  v8f acc00 = {0.f,0.f,0.f,0.f,0.f,0.f,0.f,0.f};
  v8f acc01 = acc00, acc10 = acc00, acc11 = acc00;
#pragma unroll 2
  for (int k0 = 0; k0 < K; k0 += 32) {
    v16h a0 = frag_load_f32(A, K, row0 + lm, k0, lane);
    v16h a1 = frag_load_f32(A, K, row0 + 16 + lm, k0, lane);
    v16h b0 = frag_load_f32(B, K, col0 + lm, k0, lane);
    v16h b1 = frag_load_f32(B, K, col0 + 16 + lm, k0, lane);
    acc00 = __builtin_amdgcn_wmma_f32_16x16x32_f16(false, a0, false, b0, (short)0, acc00, false, false);
    acc01 = __builtin_amdgcn_wmma_f32_16x16x32_f16(false, a0, false, b1, (short)0, acc01, false, false);
    acc10 = __builtin_amdgcn_wmma_f32_16x16x32_f16(false, a1, false, b0, (short)0, acc10, false, false);
    acc11 = __builtin_amdgcn_wmma_f32_16x16x32_f16(false, a1, false, b1, (short)0, acc11, false, false);
  }
  // C/D layout (ISA 7.12.2): VGPR r -> row = r + (lane<16?0:8), col = lane%16
  int rb = (lane < 16) ? 0 : 8;
  int ca = col0 + lm, cb = col0 + 16 + lm;
  float cba = 0.f, cbb = 0.f;
  if (biasMode == 1) { cba = bias[ca]; cbb = bias[cb]; }
#pragma unroll
  for (int r = 0; r < 8; ++r) {
    int ra = row0 + rb + r;
    int rbr = row0 + 16 + rb + r;
    float ba = cba, bb2 = cba, bc = cbb, bd = cbb;
    if (biasMode == 2) {
      float v0 = bias[ra], v1 = bias[rbr];
      ba = v0; bb2 = v1; bc = v0; bd = v1;
    }
    C[(long long)ra  * N + ca] = acc00[r] + ba;
    C[(long long)rbr * N + ca] = acc10[r] + bb2;
    C[(long long)ra  * N + cb] = acc01[r] + bc;
    C[(long long)rbr * N + cb] = acc11[r] + bd;
  }
}

// ---------------------------------------------------------------------------
// Small helper kernels
// ---------------------------------------------------------------------------
__global__ void extract_xyz_kernel(const float* __restrict__ x, float* __restrict__ xyz,
                                   long long total) {
  long long i = (long long)blockIdx.x * TPB + threadIdx.x;
  if (i >= total) return;
  long long p = i / 3; int c = (int)(i % 3);
  xyz[i] = x[p * 6 + c];
}

// out[r,o] = sum_c in[r*inStride+inOff+c] * w[o*K+c]   (tiny K)
__global__ void smallk_gemm_kernel(const float* __restrict__ in, int inStride, int inOff,
                                   const float* __restrict__ w, int K,
                                   float* __restrict__ out, long long rows, int O) {
  long long idx = (long long)blockIdx.x * TPB + threadIdx.x;
  if (idx >= rows * (long long)O) return;
  long long r = idx / O; int o = (int)(idx % O);
  float s = 0.f;
  for (int c = 0; c < K; ++c) s += in[r * inStride + inOff + c] * w[o * K + c];
  out[r * O + o] = s;
}

// per-column mean / inv-std over M rows
__global__ void colstats_kernel(const float* __restrict__ X, long long M, int N,
                                float* __restrict__ mean, float* __restrict__ istd) {
  __shared__ float s1[TPB], s2[TPB];
  int j = blockIdx.x, t = threadIdx.x;
  float a = 0.f, b = 0.f;
  for (long long r = t; r < M; r += TPB) {
    float v = X[r * N + j]; a += v; b += v * v;
  }
  s1[t] = a; s2[t] = b; __syncthreads();
  for (int s = TPB / 2; s > 0; s >>= 1) {
    if (t < s) { s1[t] += s1[t + s]; s2[t] += s2[t + s]; }
    __syncthreads();
  }
  if (t == 0) {
    float m = s1[0] / (float)M;
    float var = s2[0] / (float)M - m * m;
    mean[j] = m;
    istd[j] = rsqrtf(fmaxf(var, 0.f) + 1e-5f);
  }
}

// act: 0 none, 1 relu, 2 leaky(0.2)
__global__ void bn_apply_kernel(float* __restrict__ X, long long total, int N,
                                const float* __restrict__ mean, const float* __restrict__ istd,
                                const float* __restrict__ g, const float* __restrict__ b,
                                int act) {
  long long i = (long long)blockIdx.x * TPB + threadIdx.x;
  if (i >= total) return;
  int j = (int)(i % N);
  float y = (X[i] - mean[j]) * istd[j] * g[j] + b[j];
  if (act == 1) y = fmaxf(y, 0.f);
  else if (act == 2) y = (y > 0.f) ? y : 0.2f * y;
  X[i] = y;
}

// Farthest point sampling: one block per batch (matches lax.scan semantics:
// record current 'far', then update dists/argmax; first-index tie-break).
__global__ void fps_kernel(const float* __restrict__ xyz, int N, int npoint,
                           int* __restrict__ out) {
  __shared__ float dists[2048];
  __shared__ float rv[TPB];
  __shared__ int ri[TPB];
  int b = blockIdx.x, t = threadIdx.x;
  const float* p = xyz + (long long)b * N * 3;
  for (int i = t; i < N; i += TPB) dists[i] = 1e10f;
  __syncthreads();
  int far = 0;
  for (int it = 0; it < npoint; ++it) {
    if (t == 0) out[b * npoint + it] = far;
    float cx = p[far * 3 + 0], cy = p[far * 3 + 1], cz = p[far * 3 + 2];
    float best = -1.f; int bi = 0x7fffffff;
    for (int i = t; i < N; i += TPB) {
      float dx = p[i * 3 + 0] - cx, dy = p[i * 3 + 1] - cy, dz = p[i * 3 + 2] - cz;
      float d = dx * dx + dy * dy + dz * dz;
      float m = fminf(dists[i], d);
      dists[i] = m;
      if (m > best) { best = m; bi = i; }
    }
    rv[t] = best; ri[t] = bi;
    __syncthreads();
    for (int s = TPB / 2; s > 0; s >>= 1) {
      if (t < s) {
        if (rv[t + s] > rv[t] || (rv[t + s] == rv[t] && ri[t + s] < ri[t])) {
          rv[t] = rv[t + s]; ri[t] = ri[t + s];
        }
      }
      __syncthreads();
    }
    far = ri[0];
    __syncthreads();
  }
}

__global__ void gather_xyz_kernel(const float* __restrict__ xyz, const int* __restrict__ fidx,
                                  float* __restrict__ nxyz, int S, int N, int B) {
  int idx = blockIdx.x * TPB + threadIdx.x;
  if (idx >= B * S) return;
  int b = idx / S;
  int fi = fidx[idx];
  long long src = ((long long)b * N + fi) * 3;
  long long dst = (long long)idx * 3;
  nxyz[dst + 0] = xyz[src + 0];
  nxyz[dst + 1] = xyz[src + 1];
  nxyz[dst + 2] = xyz[src + 2];
}

// k-NN by iterative selection: one block per query point.
__global__ void knn_kernel(const float* __restrict__ q, const float* __restrict__ p,
                           int S, int N, int k, int* __restrict__ out) {
  __shared__ float d[2048];
  __shared__ float rv[TPB];
  __shared__ int ri[TPB];
  int bs = blockIdx.x;          // b*S + s
  int b = bs / S;
  int t = threadIdx.x;
  const float* pb = p + (long long)b * N * 3;
  float qx = q[(long long)bs * 3 + 0];
  float qy = q[(long long)bs * 3 + 1];
  float qz = q[(long long)bs * 3 + 2];
  for (int i = t; i < N; i += TPB) {
    float dx = pb[i * 3 + 0] - qx, dy = pb[i * 3 + 1] - qy, dz = pb[i * 3 + 2] - qz;
    d[i] = dx * dx + dy * dy + dz * dz;
  }
  __syncthreads();
  for (int j = 0; j < k; ++j) {
    float best = 1e30f; int bi = 0x7fffffff;
    for (int i = t; i < N; i += TPB) {
      float v = d[i];
      if (v < best || (v == best && i < bi)) { best = v; bi = i; }
    }
    rv[t] = best; ri[t] = bi;
    __syncthreads();
    for (int s = TPB / 2; s > 0; s >>= 1) {
      if (t < s) {
        if (rv[t + s] < rv[t] || (rv[t + s] == rv[t] && ri[t + s] < ri[t])) {
          rv[t] = rv[t + s]; ri[t] = ri[t + s];
        }
      }
      __syncthreads();
    }
    if (t == 0) { out[(long long)bs * k + j] = ri[0]; d[ri[0]] = 1e30f; }
    __syncthreads();
  }
}

// Build grouped features: G[(g*k+kk), c] = P[b,nbr,c]-P[b,ctr,c];  [C+c] = P[b,ctr,c]
__global__ void group_kernel(const float* __restrict__ P, const int* __restrict__ knn,
                             const int* __restrict__ fidx, float* __restrict__ G,
                             int S, int N, int C, int k, long long total) {
  long long idx = (long long)blockIdx.x * TPB + threadIdx.x;
  if (idx >= total) return;
  int c = (int)(idx % C);
  long long t = idx / C;
  int kk = (int)(t % k);
  long long g = t / k;          // b*S + s
  int b = (int)(g / S);
  int j = knn[g * k + kk];
  int fi = fidx[g];
  float pv = P[((long long)b * N + j) * C + c];
  float cv = P[((long long)b * N + fi) * C + c];
  long long row = (g * k + kk) * (long long)(2 * C);
  G[row + c] = pv - cv;
  G[row + C + c] = cv;
}

// F[g,o] = max_k H[(g*k+kk), o]
__global__ void maxk_kernel(const float* __restrict__ H, float* __restrict__ F,
                            long long groups, int k, int C) {
  long long idx = (long long)blockIdx.x * TPB + threadIdx.x;
  if (idx >= groups * (long long)C) return;
  int o = (int)(idx % C);
  long long g = idx / C;
  float best = -1e30f;
  for (int kk = 0; kk < k; ++kk) best = fmaxf(best, H[(g * k + kk) * (long long)C + o]);
  F[g * C + o] = best;
}

// row-wise softmax, cols == blockDim == 256, in place
__global__ void softmax_rows_kernel(float* __restrict__ E, int cols) {
  __shared__ float red[TPB];
  long long r = blockIdx.x;
  int t = threadIdx.x;
  float v = E[r * cols + t];
  red[t] = v; __syncthreads();
  for (int s = TPB / 2; s > 0; s >>= 1) { if (t < s) red[t] = fmaxf(red[t], red[t + s]); __syncthreads(); }
  float mx = red[0]; __syncthreads();
  float e = __expf(v - mx);
  red[t] = e; __syncthreads();
  for (int s = TPB / 2; s > 0; s >>= 1) { if (t < s) red[t] += red[t + s]; __syncthreads(); }
  float sm = red[0];
  E[r * cols + t] = e / sm;
}

// cs[b,m] = sum_n E[b,n,m]
__global__ void colsum_kernel(const float* __restrict__ E, float* __restrict__ cs,
                              int B, int n) {
  int idx = blockIdx.x * TPB + threadIdx.x;
  if (idx >= B * n) return;
  int m = idx % n, b = idx / n;
  float s = 0.f;
  for (int nn = 0; nn < n; ++nn) s += E[((long long)b * n + nn) * n + m];
  cs[idx] = s;
}

// ET[b,m,n] = E[b,n,m] / (1e-9 + cs[b,m])  (normalize + transpose)
__global__ void attn_norm_t_kernel(const float* __restrict__ E, const float* __restrict__ cs,
                                   float* __restrict__ ET, int B, int n) {
  long long idx = (long long)blockIdx.x * TPB + threadIdx.x;
  long long total = (long long)B * n * n;
  if (idx >= total) return;
  int nn = (int)(idx % n);
  long long t = idx / n;
  int m = (int)(t % n);
  int b = (int)(t / n);
  ET[idx] = E[((long long)b * n + nn) * n + m] / (1e-9f + cs[b * n + m]);
}

__global__ void add_kernel(float* __restrict__ y, const float* __restrict__ s, long long total) {
  long long i = (long long)blockIdx.x * TPB + threadIdx.x;
  if (i >= total) return;
  y[i] += s[i];
}

// D[b,n,c] = X[b,n,c] - XRT[b,c,n]
__global__ void sub_t_kernel(const float* __restrict__ X, const float* __restrict__ XRT,
                             float* __restrict__ D, int B, int n) {
  long long idx = (long long)blockIdx.x * TPB + threadIdx.x;
  long long total = (long long)B * n * n;
  if (idx >= total) return;
  int c = (int)(idx % n);
  long long t = idx / n;
  int nn = (int)(t % n);
  int b = (int)(t / n);
  D[idx] = X[idx] - XRT[((long long)b * n + c) * n + nn];
}

__global__ void copy_cols_kernel(const float* __restrict__ S, float* __restrict__ D,
                                 long long rows, int sc, int dc, int off) {
  long long idx = (long long)blockIdx.x * TPB + threadIdx.x;
  if (idx >= rows * (long long)sc) return;
  int c = (int)(idx % sc);
  long long r = idx / sc;
  D[r * dc + off + c] = S[idx];
}

// P[b,j] = max_n F[b,n,j]
__global__ void maxpool_kernel(const float* __restrict__ F, float* __restrict__ P,
                               int B, int n, int C) {
  int idx = blockIdx.x * TPB + threadIdx.x;
  if (idx >= B * C) return;
  int j = idx % C, b = idx / C;
  float best = -1e30f;
  for (int nn = 0; nn < n; ++nn) best = fmaxf(best, F[((long long)b * n + nn) * C + j]);
  P[idx] = best;
}

// out[m,j] = sum_k in[m,k]*w[j,k] (+bias[j])   (tiny M head layers)
__global__ void linear_kernel(const float* __restrict__ in, const float* __restrict__ w,
                              const float* __restrict__ bias, float* __restrict__ out,
                              int M, int K, int Nout) {
  int idx = blockIdx.x * TPB + threadIdx.x;
  if (idx >= M * Nout) return;
  int j = idx % Nout, m = idx / Nout;
  float s = (bias != nullptr) ? bias[j] : 0.f;
  for (int k = 0; k < K; ++k) s += in[m * K + k] * w[j * K + k];
  out[idx] = s;
}

// BN over axis 0 (M=32 samples): one wave per feature column.
__global__ void bn_rows_kernel(float* __restrict__ X, int N, const float* __restrict__ g,
                               const float* __restrict__ b, int act) {
  int j = blockIdx.x, m = threadIdx.x;   // blockDim == 32
  float v = X[m * N + j];
  float s = v;
  for (int o = 16; o > 0; o >>= 1) s += __shfl_xor(s, o, 32);
  float mean = s / 32.f;
  float d = v - mean;
  float q = d * d;
  for (int o = 16; o > 0; o >>= 1) q += __shfl_xor(q, o, 32);
  float var = q / 32.f;
  float y = d * rsqrtf(var + 1e-5f) * g[j] + b[j];
  if (act == 1) y = fmaxf(y, 0.f);
  X[m * N + j] = y;
}

// ---------------------------------------------------------------------------
// Input index map: jax pytree flattening of setup_inputs() (dicts sorted by key)
// ---------------------------------------------------------------------------
enum {
  IN_X = 0,
  IN_BN1_B, IN_BN1_G, IN_BN2_B, IN_BN2_G, IN_BN6_B, IN_BN6_G, IN_BN7_B, IN_BN7_G,
  IN_CONV1_W, IN_CONV2_W, IN_FUSE_B, IN_FUSE_G, IN_FUSE_W, IN_L1_W,
  IN_L2_B, IN_L2_W, IN_L3_B, IN_L3_W,
  IN_PCT_BN1_B, IN_PCT_BN1_G, IN_PCT_C1_W, IN_PCT_POS_W,
  IN_SA0 = 23,   // per SA (x4): bn_b, bn_g, qk_w, t_b, t_w, v_b, v_w
  IN_TD0 = 51,   // b1, b2, c1, c2, g1, g2
  IN_TD1 = 57
};

static inline long long cdiv(long long a, long long b) { return (a + b - 1) / b; }

extern "C" void kernel_launch(void* const* d_in, const int* in_sizes, int n_in,
                              void* d_out, int out_size, void* d_ws, size_t ws_size,
                              hipStream_t stream) {
  (void)in_sizes; (void)n_in; (void)out_size; (void)ws_size;
  const int B = 32, N0 = 2048, S0 = 512, S1 = 256, KNB = 32;

  const float* X       = (const float*)d_in[IN_X];
  const float* bn1_b   = (const float*)d_in[IN_BN1_B];
  const float* bn1_g   = (const float*)d_in[IN_BN1_G];
  const float* bn2_b   = (const float*)d_in[IN_BN2_B];
  const float* bn2_g   = (const float*)d_in[IN_BN2_G];
  const float* bn6_b   = (const float*)d_in[IN_BN6_B];
  const float* bn6_g   = (const float*)d_in[IN_BN6_G];
  const float* bn7_b   = (const float*)d_in[IN_BN7_B];
  const float* bn7_g   = (const float*)d_in[IN_BN7_G];
  const float* conv1_w = (const float*)d_in[IN_CONV1_W];
  const float* conv2_w = (const float*)d_in[IN_CONV2_W];
  const float* fuse_b  = (const float*)d_in[IN_FUSE_B];
  const float* fuse_g  = (const float*)d_in[IN_FUSE_G];
  const float* fuse_w  = (const float*)d_in[IN_FUSE_W];
  const float* l1_w    = (const float*)d_in[IN_L1_W];
  const float* l2_b    = (const float*)d_in[IN_L2_B];
  const float* l2_w    = (const float*)d_in[IN_L2_W];
  const float* l3_b    = (const float*)d_in[IN_L3_B];
  const float* l3_w    = (const float*)d_in[IN_L3_W];
  const float* pct_bn1_b = (const float*)d_in[IN_PCT_BN1_B];
  const float* pct_bn1_g = (const float*)d_in[IN_PCT_BN1_G];
  const float* pct_c1_w  = (const float*)d_in[IN_PCT_C1_W];
  const float* pct_pos_w = (const float*)d_in[IN_PCT_POS_W];
  const float* td0_b1 = (const float*)d_in[IN_TD0 + 0];
  const float* td0_b2 = (const float*)d_in[IN_TD0 + 1];
  const float* td0_c1 = (const float*)d_in[IN_TD0 + 2];
  const float* td0_c2 = (const float*)d_in[IN_TD0 + 3];
  const float* td0_g1 = (const float*)d_in[IN_TD0 + 4];
  const float* td0_g2 = (const float*)d_in[IN_TD0 + 5];
  const float* td1_b1 = (const float*)d_in[IN_TD1 + 0];
  const float* td1_b2 = (const float*)d_in[IN_TD1 + 1];
  const float* td1_c1 = (const float*)d_in[IN_TD1 + 2];
  const float* td1_c2 = (const float*)d_in[IN_TD1 + 3];
  const float* td1_g1 = (const float*)d_in[IN_TD1 + 4];
  const float* td1_g2 = (const float*)d_in[IN_TD1 + 5];

  // workspace carving (256B aligned slots)
  float* wsf = (float*)d_ws;
  size_t off = 0;
  auto alloc = [&](size_t n) { float* p = wsf + off; off += (n + 63) & ~(size_t)63; return p; };
  float* xyz0   = alloc((size_t)B * N0 * 3);
  float* feats  = alloc((size_t)B * N0 * 64);      // conv1 out
  float* feats2 = alloc((size_t)B * N0 * 64);      // conv2 out = td0 points
  int*   fidx0  = (int*)alloc((size_t)B * S0);
  float* nxyz0  = alloc((size_t)B * S0 * 3);
  int*   knn0   = (int*)alloc((size_t)B * S0 * KNB);
  float* Gbuf   = alloc((size_t)B * S0 * KNB * 128);   // 67.1M floats, reused by td1
  float* Hbuf   = alloc((size_t)B * S0 * KNB * 128);   // 67.1M floats, reused by td1
  float* f0     = alloc((size_t)B * S0 * 128);
  int*   fidx1  = (int*)alloc((size_t)B * S1);
  float* nxyz1  = alloc((size_t)B * S1 * 3);
  int*   knn1   = (int*)alloc((size_t)B * S1 * KNB);
  float* f1     = alloc((size_t)B * S1 * 256);
  float* pe     = alloc((size_t)B * S1 * 256);
  float* xbuf   = alloc((size_t)B * S1 * 256);
  float* xk     = alloc((size_t)B * S1 * 64);
  float* energy = alloc((size_t)B * S1 * S1);
  float* attnT  = alloc((size_t)B * S1 * S1);
  float* csum   = alloc((size_t)B * S1);
  float* vT     = alloc((size_t)B * S1 * 256);
  float* xrT    = alloc((size_t)B * S1 * 256);
  float* Dbuf   = alloc((size_t)B * S1 * 256);
  float* Hsa    = alloc((size_t)B * S1 * 256);
  float* fusein = alloc((size_t)B * S1 * 1280);
  float* fuseout= alloc((size_t)B * S1 * 1024);
  float* pooled = alloc((size_t)B * 1024);
  float* h1b    = alloc((size_t)B * 512);
  float* h2b    = alloc((size_t)B * 256);
  float* meanb  = alloc(2048);
  float* istdb  = alloc(2048);

  auto gemm = [&](const float* A, const float* Bm, float* C, int M, int N, int K,
                  long long sA, long long sB, long long sC, int batches,
                  const float* bias, int biasMode) {
    long long tiles = (long long)(M / 32) * (N / 32);
    dim3 g((unsigned)cdiv(tiles, 8), 1, (unsigned)batches);
    wmma_gemm_kernel<<<g, TPB, 0, stream>>>(A, Bm, C, M, N, K, sA, sB, sC, bias, biasMode);
  };
  auto bn = [&](float* Xb, long long M, int N, const float* g, const float* b, int act) {
    colstats_kernel<<<dim3(N), TPB, 0, stream>>>(Xb, M, N, meanb, istdb);
    long long tot = M * N;
    bn_apply_kernel<<<dim3((unsigned)cdiv(tot, TPB)), TPB, 0, stream>>>(
        Xb, tot, N, meanb, istdb, g, b, act);
  };

  // ---- stem -------------------------------------------------------------
  {
    long long tot = (long long)B * N0 * 3;
    extract_xyz_kernel<<<dim3((unsigned)cdiv(tot, TPB)), TPB, 0, stream>>>(X, xyz0, tot);
  }
  smallk_gemm_kernel<<<dim3((unsigned)cdiv((long long)B * N0 * 64, TPB)), TPB, 0, stream>>>(
      X, 6, 3, conv1_w, 3, feats, (long long)B * N0, 64);
  bn(feats, (long long)B * N0, 64, bn1_g, bn1_b, 1);
  gemm(feats, conv2_w, feats2, B * N0, 64, 64, 0, 0, 0, 1, nullptr, 0);
  bn(feats2, (long long)B * N0, 64, bn2_g, bn2_b, 1);

  // ---- transition down 0: 2048 -> 512, C 64 -> 128 ----------------------
  fps_kernel<<<dim3(B), TPB, 0, stream>>>(xyz0, N0, S0, fidx0);
  gather_xyz_kernel<<<dim3((unsigned)cdiv(B * S0, TPB)), TPB, 0, stream>>>(xyz0, fidx0, nxyz0, S0, N0, B);
  knn_kernel<<<dim3(B * S0), TPB, 0, stream>>>(nxyz0, xyz0, S0, N0, KNB, knn0);
  {
    long long tot = (long long)B * S0 * KNB * 64;
    group_kernel<<<dim3((unsigned)cdiv(tot, TPB)), TPB, 0, stream>>>(
        feats2, knn0, fidx0, Gbuf, S0, N0, 64, KNB, tot);
  }
  gemm(Gbuf, td0_c1, Hbuf, B * S0 * KNB, 128, 128, 0, 0, 0, 1, nullptr, 0);
  bn(Hbuf, (long long)B * S0 * KNB, 128, td0_g1, td0_b1, 1);
  gemm(Hbuf, td0_c2, Gbuf, B * S0 * KNB, 128, 128, 0, 0, 0, 1, nullptr, 0);
  bn(Gbuf, (long long)B * S0 * KNB, 128, td0_g2, td0_b2, 1);
  maxk_kernel<<<dim3((unsigned)cdiv((long long)B * S0 * 128, TPB)), TPB, 0, stream>>>(
      Gbuf, f0, (long long)B * S0, KNB, 128);

  // ---- transition down 1: 512 -> 256, C 128 -> 256 ----------------------
  fps_kernel<<<dim3(B), TPB, 0, stream>>>(nxyz0, S0, S1, fidx1);
  gather_xyz_kernel<<<dim3((unsigned)cdiv(B * S1, TPB)), TPB, 0, stream>>>(nxyz0, fidx1, nxyz1, S1, S0, B);
  knn_kernel<<<dim3(B * S1), TPB, 0, stream>>>(nxyz1, nxyz0, S1, S0, KNB, knn1);
  {
    long long tot = (long long)B * S1 * KNB * 128;
    group_kernel<<<dim3((unsigned)cdiv(tot, TPB)), TPB, 0, stream>>>(
        f0, knn1, fidx1, Gbuf, S1, S0, 128, KNB, tot);
  }
  gemm(Gbuf, td1_c1, Hbuf, B * S1 * KNB, 256, 256, 0, 0, 0, 1, nullptr, 0);
  bn(Hbuf, (long long)B * S1 * KNB, 256, td1_g1, td1_b1, 1);
  gemm(Hbuf, td1_c2, Gbuf, B * S1 * KNB, 256, 256, 0, 0, 0, 1, nullptr, 0);
  bn(Gbuf, (long long)B * S1 * KNB, 256, td1_g2, td1_b2, 1);
  maxk_kernel<<<dim3((unsigned)cdiv((long long)B * S1 * 256, TPB)), TPB, 0, stream>>>(
      Gbuf, f1, (long long)B * S1, KNB, 256);

  // ---- PCT --------------------------------------------------------------
  const long long MP = (long long)B * S1;        // 8192 rows
  smallk_gemm_kernel<<<dim3((unsigned)cdiv(MP * 256, TPB)), TPB, 0, stream>>>(
      nxyz1, 3, 0, pct_pos_w, 3, pe, MP, 256);
  gemm(f1, pct_c1_w, xbuf, (int)MP, 256, 256, 0, 0, 0, 1, nullptr, 0);
  bn(xbuf, MP, 256, pct_bn1_g, pct_bn1_b, 1);

  const long long TOTX = MP * 256;
  for (int i = 0; i < 4; ++i) {
    const float* sa_bn_b = (const float*)d_in[IN_SA0 + 7 * i + 0];
    const float* sa_bn_g = (const float*)d_in[IN_SA0 + 7 * i + 1];
    const float* sa_qk_w = (const float*)d_in[IN_SA0 + 7 * i + 2];
    const float* sa_t_b  = (const float*)d_in[IN_SA0 + 7 * i + 3];
    const float* sa_t_w  = (const float*)d_in[IN_SA0 + 7 * i + 4];
    const float* sa_v_b  = (const float*)d_in[IN_SA0 + 7 * i + 5];
    const float* sa_v_w  = (const float*)d_in[IN_SA0 + 7 * i + 6];

    add_kernel<<<dim3((unsigned)cdiv(TOTX, TPB)), TPB, 0, stream>>>(xbuf, pe, TOTX);
    // xk[b,n,c4] ; energy[b,n,m] = sum_c xk[b,n,c]*xk[b,m,c]
    gemm(xbuf, sa_qk_w, xk, (int)MP, 64, 256, 0, 0, 0, 1, nullptr, 0);
    gemm(xk, xk, energy, S1, S1, 64, (long long)S1 * 64, (long long)S1 * 64,
         (long long)S1 * S1, B, nullptr, 0);
    // vT[b,c,n] = sum_k v_w[c,k]*x[b,n,k] + v_b[c]
    gemm(sa_v_w, xbuf, vT, 256, S1, 256, 0, (long long)S1 * 256, (long long)S1 * 256,
         B, sa_v_b, 2);
    softmax_rows_kernel<<<dim3((unsigned)(B * S1)), TPB, 0, stream>>>(energy, S1);
    colsum_kernel<<<dim3((unsigned)cdiv(B * S1, TPB)), TPB, 0, stream>>>(energy, csum, B, S1);
    attn_norm_t_kernel<<<dim3((unsigned)cdiv((long long)B * S1 * S1, TPB)), TPB, 0, stream>>>(
        energy, csum, attnT, B, S1);
    // xrT[b,c,m] = sum_n vT[b,c,n]*attnT[b,m,n]
    gemm(vT, attnT, xrT, 256, S1, S1, (long long)S1 * 256, (long long)S1 * S1,
         (long long)S1 * 256, B, nullptr, 0);
    sub_t_kernel<<<dim3((unsigned)cdiv(TOTX, TPB)), TPB, 0, stream>>>(xbuf, xrT, Dbuf, B, S1);
    gemm(Dbuf, sa_t_w, Hsa, (int)MP, 256, 256, 0, 0, 0, 1, sa_t_b, 1);
    bn(Hsa, MP, 256, sa_bn_g, sa_bn_b, 1);
    add_kernel<<<dim3((unsigned)cdiv(TOTX, TPB)), TPB, 0, stream>>>(xbuf, Hsa, TOTX);
    copy_cols_kernel<<<dim3((unsigned)cdiv(TOTX, TPB)), TPB, 0, stream>>>(
        xbuf, fusein, MP, 256, 1280, i * 256);
  }
  copy_cols_kernel<<<dim3((unsigned)cdiv(TOTX, TPB)), TPB, 0, stream>>>(
      f1, fusein, MP, 256, 1280, 1024);

  // ---- fuse + pool ------------------------------------------------------
  gemm(fusein, fuse_w, fuseout, (int)MP, 1024, 1280, 0, 0, 0, 1, nullptr, 0);
  bn(fuseout, MP, 1024, fuse_g, fuse_b, 2);
  maxpool_kernel<<<dim3((unsigned)cdiv(B * 1024, TPB)), TPB, 0, stream>>>(
      fuseout, pooled, B, S1, 1024);

  // ---- head -------------------------------------------------------------
  linear_kernel<<<dim3((unsigned)cdiv(B * 512, TPB)), TPB, 0, stream>>>(
      pooled, l1_w, nullptr, h1b, B, 1024, 512);
  bn_rows_kernel<<<dim3(512), 32, 0, stream>>>(h1b, 512, bn6_g, bn6_b, 1);
  linear_kernel<<<dim3((unsigned)cdiv(B * 256, TPB)), TPB, 0, stream>>>(
      h1b, l2_w, l2_b, h2b, B, 512, 256);
  bn_rows_kernel<<<dim3(256), 32, 0, stream>>>(h2b, 256, bn7_g, bn7_b, 1);
  linear_kernel<<<dim3((unsigned)cdiv(B * 40, TPB)), TPB, 0, stream>>>(
      h2b, l3_w, l3_b, (float*)d_out, B, 256, 40);
}
